// SharedSparseOffsetDict_24180665876983
// MI455X (gfx1250) — matrix operand order repaired and checked by reading, not compiled, not run on hardware
//
#include <hip/hip_runtime.h>

// ---------------------------------------------------------------------------
// SharedSparseOffsetDict for MI455X (gfx1250, wave32, WMMA + TDM)
//   coeffs = x @ W_enc^T   (M=4096, N=16384, K=1024)  -> fused top-8 per token
//   offset = sum_k val_k * W_dict[idx_k]
//   loss   = sum |vals| / (4096*16384)
// ---------------------------------------------------------------------------

typedef unsigned short u16;
typedef __bf16  v8bf  __attribute__((ext_vector_type(8)));
typedef __bf16  v16bf __attribute__((ext_vector_type(16)));
typedef float   v8f   __attribute__((ext_vector_type(8)));
typedef unsigned int u32x4 __attribute__((ext_vector_type(4)));
typedef int          i32x4 __attribute__((ext_vector_type(4)));
typedef int          i32x8 __attribute__((ext_vector_type(8)));

#define BT_TOKENS 4096
#define D_MODEL   1024
#define DICT_SZ   16384
#define TOPK      8

// A tile staged by TDM: 32 rows x (2048B data + 2x16B pad) = 2080B row stride.
// Pad keeps b128 frags 16B-aligned and gives 520-dword row stride (2-way banks).
#define A_ROW_BYTES 2080
#define A_TILE_BYTES (32 * A_ROW_BYTES)

#define WMMA_BF16(a, b, c) \
  __builtin_amdgcn_wmma_f32_16x16x32_bf16(false, (a), false, (b), (short)0, (c), false, false)

// fp32 -> bf16 round-to-nearest-even
__device__ __forceinline__ u16 f2bf(float f) {
  unsigned u = __float_as_uint(f);
  u += 0x7FFFu + ((u >> 16) & 1u);
  return (u16)(u >> 16);
}

// ---------------------------------------------------------------------------
// Kernel 1: fp32 -> bf16 conversion (4 elems / thread, exact grids)
// ---------------------------------------------------------------------------
__global__ __launch_bounds__(256) void cvt_bf16_kernel(const float* __restrict__ src,
                                                       u16* __restrict__ dst, int n) {
  int i = (blockIdx.x * blockDim.x + threadIdx.x) * 4;
  if (i >= n) return;
  float4 f = *(const float4*)(src + i);
  union { u16 u[4]; unsigned long long ll; } t;
  t.u[0] = f2bf(f.x); t.u[1] = f2bf(f.y); t.u[2] = f2bf(f.z); t.u[3] = f2bf(f.w);
  *(unsigned long long*)(dst + i) = t.ll;
}

// ---------------------------------------------------------------------------
// Kernel 2: fused WMMA GEMM + per-token top-8
//   grid = 128 blocks (32-token M tile each), block = 128 threads = 4 waves
//   A tile (32x1024 bf16) staged once per block into LDS by the TDM;
//   wave w streams N stripe [w*4096, (w+1)*4096) in 32-column tiles from L2.
// ---------------------------------------------------------------------------
__global__ __launch_bounds__(128) void encode_topk_kernel(
    const u16* __restrict__ xb, const u16* __restrict__ wb,
    float* __restrict__ tv_out, int* __restrict__ ti_out,
    float* __restrict__ rowabs) {
  __shared__ __align__(16) char sA[A_TILE_BYTES];      // 66,560 B
  __shared__ float sC[4][32 * 36];                     // 18,432 B
  __shared__ float mV[3][32][TOPK];                    //  3,072 B
  __shared__ int   mI[3][32][TOPK];                    //  3,072 B

  const int tid  = threadIdx.x;
  const int w    = tid >> 5;        // wave 0..3
  const int lane = tid & 31;
  const int hl   = lane >> 4;       // lane half (0/1)
  const int r    = lane & 15;
  const int am   = blockIdx.x * 32; // first token row of this block

  // ---- TDM: DMA the 32x1024 bf16 A tile (row-padded) into LDS --------------
  if (w == 0) {
    const unsigned lds_base = (unsigned)(size_t)&sA[0];
    const unsigned long long ga =
        (unsigned long long)(size_t)xb + (unsigned long long)am * (D_MODEL * 2ull);

    u32x4 g0;
    g0.x = 1u;                                   // count=1, user descriptor
    g0.y = lds_base;                             // lds_addr
    g0.z = (unsigned)(ga & 0xFFFFFFFFull);       // global_addr[31:0]
    g0.w = (unsigned)((ga >> 32) & 0x01FFFFFFull) | (2u << 30);  // [56:32] | type=2

    i32x8 g1;
    g1[0] = (1 << 16)      // data_size = 2 bytes
          | (1 << 20)      // pad_enable
          | (7 << 22)      // pad_interval: every 256 dwords (1024 B)
          | (3 << 25);     // pad_amount: 4 dwords (16 B)
    g1[1] = (int)((D_MODEL & 0xFFFFu) << 16);    // tensor_dim0[15:0] in [63:48]
    g1[2] = (int)((BT_TOKENS & 0xFFFFu) << 16);  // dim0 hi=0 | tensor_dim1[15:0]
    g1[3] = (int)((D_MODEL & 0xFFFFu) << 16);    // dim1 hi=0 | tile_dim0=1024
    g1[4] = 32;                                  // tile_dim1=32, tile_dim2=0
    g1[5] = D_MODEL;                             // tensor_dim0_stride = 1024
    g1[6] = 0;
    g1[7] = 0;

    i32x4 gz = {0, 0, 0, 0};
#if defined(__clang_major__) && (__clang_major__ >= 23)
    i32x8 gz8 = {0, 0, 0, 0, 0, 0, 0, 0};
    __builtin_amdgcn_tensor_load_to_lds(g0, g1, gz, gz, gz8, 0);
#else
    __builtin_amdgcn_tensor_load_to_lds(g0, g1, gz, gz, 0);
#endif
    __builtin_amdgcn_s_wait_tensorcnt((short)0);
  }
  __syncthreads();

  float tv[TOPK]; int ti[TOPK];
#pragma unroll
  for (int k = 0; k < TOPK; ++k) { tv[k] = -3.4e38f; ti[k] = 0; }

  float* myC = sC[w];
  const char* arow0 = sA + (size_t)r * A_ROW_BYTES;          // rows 0..15
  const char* arow1 = sA + (size_t)(16 + r) * A_ROW_BYTES;   // rows 16..31

  for (int it = 0; it < 128; ++it) {
    const int n0 = w * 4096 + it * 32;
    const size_t colB0 = (size_t)(n0 + r) * D_MODEL;
    const size_t colB1 = (size_t)(n0 + 16 + r) * D_MODEL;

    // prefetch next iteration's B columns (streaming, L2-resident operand)
    if (it < 127) {
      __builtin_prefetch(wb + colB0 + 32 * D_MODEL, 0, 1);
      __builtin_prefetch(wb + colB1 + 32 * D_MODEL, 0, 1);
    }

    v8f c00 = {}, c01 = {}, c10 = {}, c11 = {};

#pragma unroll 2
    for (int kk = 0; kk < D_MODEL; kk += 32) {
      // A frag (16x32 bf16) from LDS; +16B bump past mid-row TDM pad
      const int aoff = 2 * (kk + hl * 16) + ((kk >= 512) ? 16 : 0);
      v8bf a0lo = *(const v8bf*)(arow0 + aoff);
      v8bf a0hi = *(const v8bf*)(arow0 + aoff + 16);
      v8bf a1lo = *(const v8bf*)(arow1 + aoff);
      v8bf a1hi = *(const v8bf*)(arow1 + aoff + 16);
      v16bf a0 = __builtin_shufflevector(a0lo, a0hi, 0,1,2,3,4,5,6,7,8,9,10,11,12,13,14,15);
      v16bf a1 = __builtin_shufflevector(a1lo, a1hi, 0,1,2,3,4,5,6,7,8,9,10,11,12,13,14,15);
      // B frag (32x16 bf16): lane = column, 16 contiguous K per lane half
      v16bf b0 = *(const v16bf*)(wb + colB0 + kk + hl * 16);
      v16bf b1 = *(const v16bf*)(wb + colB1 + kk + hl * 16);

      c00 = WMMA_BF16(a0, b0, c00);
      c01 = WMMA_BF16(a0, b1, c01);
      c10 = WMMA_BF16(a1, b0, c10);
      c11 = WMMA_BF16(a1, b1, c11);
    }

    // stage 32x32 tile to LDS: C VGPR j -> row (j + hl*8), col = r
#pragma unroll
    for (int j = 0; j < 8; ++j) {
      const int mr = j + hl * 8;
      myC[mr * 36 + r]             = c00[j];
      myC[mr * 36 + 16 + r]        = c01[j];
      myC[(mr + 16) * 36 + r]      = c10[j];
      myC[(mr + 16) * 36 + 16 + r] = c11[j];
    }
    // within-wave LDS visibility (DS ops are in-order per wave; wait completion)
    asm volatile("s_wait_dscnt 0" ::: "memory");

    // lane L scans row L: bulk-load the row as 8 x b128, then register-scan
    const float4* rowp4 = (const float4*)(myC + lane * 36);
    float4 q[8];
#pragma unroll
    for (int c4 = 0; c4 < 8; ++c4) q[c4] = rowp4[c4];

#pragma unroll
    for (int c4 = 0; c4 < 8; ++c4) {
      const float vv[4] = {q[c4].x, q[c4].y, q[c4].z, q[c4].w};
#pragma unroll
      for (int e = 0; e < 4; ++e) {
        float v = vv[e];
        if (v > tv[0]) {
          tv[0] = v; ti[0] = n0 + c4 * 4 + e;
#pragma unroll
          for (int j = 0; j < TOPK - 1; ++j) {
            if (tv[j] > tv[j + 1]) {
              float tf = tv[j]; tv[j] = tv[j + 1]; tv[j + 1] = tf;
              int   tq = ti[j]; ti[j] = ti[j + 1]; ti[j + 1] = tq;
            }
          }
        }
      }
    }
  }

  // merge the 4 N-stripe partial top-8s (per row) via LDS
  __syncthreads();
  if (w > 0) {
#pragma unroll
    for (int k = 0; k < TOPK; ++k) { mV[w - 1][lane][k] = tv[k]; mI[w - 1][lane][k] = ti[k]; }
  }
  __syncthreads();
  if (w == 0) {
    for (int s = 0; s < 3; ++s) {
#pragma unroll
      for (int k = 0; k < TOPK; ++k) {
        float v = mV[s][lane][k]; int n = mI[s][lane][k];
        if (v > tv[0]) {
          tv[0] = v; ti[0] = n;
#pragma unroll
          for (int j = 0; j < TOPK - 1; ++j) {
            if (tv[j] > tv[j + 1]) {
              float tf = tv[j]; tv[j] = tv[j + 1]; tv[j + 1] = tf;
              int   tq = ti[j]; ti[j] = ti[j + 1]; ti[j + 1] = tq;
            }
          }
        }
      }
    }
    const int token = am + lane;
    float sum = 0.f;
#pragma unroll
    for (int k = 0; k < TOPK; ++k) {
      tv_out[token * TOPK + k] = tv[k];
      ti_out[token * TOPK + k] = ti[k];
      sum += fabsf(tv[k]);
    }
    rowabs[token] = sum;
  }
}

// ---------------------------------------------------------------------------
// Kernel 3: offset[t,:] = sum_k val_k * W_dict[idx_k,:]   (1 block / token)
// ---------------------------------------------------------------------------
__global__ __launch_bounds__(256) void decode_kernel(const float* __restrict__ Wd,
                                                     const float* __restrict__ tv,
                                                     const int* __restrict__ ti,
                                                     float* __restrict__ out) {
  const int t = blockIdx.x;
  __shared__ float sv[TOPK];
  __shared__ int   si[TOPK];
  if (threadIdx.x < TOPK) sv[threadIdx.x] = tv[t * TOPK + threadIdx.x];
  else if (threadIdx.x < 2 * TOPK) si[threadIdx.x - TOPK] = ti[t * TOPK + threadIdx.x - TOPK];
  __syncthreads();

  const int c = threadIdx.x;  // 4-wide column group, 256*4 = 1024 cols
  float4 acc = make_float4(0.f, 0.f, 0.f, 0.f);
#pragma unroll
  for (int k = 0; k < TOPK; ++k) {
    const float4* wr = (const float4*)(Wd + (size_t)si[k] * D_MODEL);
    float4 q = wr[c];
    float v = sv[k];
    acc.x += v * q.x; acc.y += v * q.y; acc.z += v * q.z; acc.w += v * q.w;
  }
  ((float4*)(out + (size_t)t * D_MODEL))[c] = acc;
}

// ---------------------------------------------------------------------------
// Kernel 4: deterministic loss reduction
// ---------------------------------------------------------------------------
__global__ __launch_bounds__(256) void loss_kernel(const float* __restrict__ rowabs,
                                                   float* __restrict__ out) {
  __shared__ float red[256];
  float s = 0.f;
  for (int i = threadIdx.x; i < BT_TOKENS; i += 256) s += rowabs[i];
  red[threadIdx.x] = s;
  __syncthreads();
  for (int o = 128; o > 0; o >>= 1) {
    if (threadIdx.x < o) red[threadIdx.x] += red[threadIdx.x + o];
    __syncthreads();
  }
  if (threadIdx.x == 0)
    out[(size_t)BT_TOKENS * D_MODEL] = red[0] * (1.0f / 67108864.0f);
}

// ---------------------------------------------------------------------------
extern "C" void kernel_launch(void* const* d_in, const int* in_sizes, int n_in,
                              void* d_out, int out_size, void* d_ws, size_t ws_size,
                              hipStream_t stream) {
  const float* x  = (const float*)d_in[0];   // [2,2048,1024]
  const float* We = (const float*)d_in[1];   // [16384,1024]
  const float* Wd = (const float*)d_in[2];   // [16384,1024]
  float* out = (float*)d_out;                // 4194304 offset + 1 loss

  char* ws = (char*)d_ws;
  u16*   xb     = (u16*)ws;                                        //  8 MB
  u16*   wb     = (u16*)(ws + (size_t)(8u << 20));                 // 32 MB
  float* tv     = (float*)(ws + (size_t)(40u << 20));              // 128 KB
  int*   ti     = (int*)  (ws + (size_t)(40u << 20) + (128u << 10));
  float* rowabs = (float*)(ws + (size_t)(40u << 20) + (256u << 10));

  const int nx = BT_TOKENS * D_MODEL;   //  4,194,304
  const int nw = DICT_SZ * D_MODEL;     // 16,777,216

  cvt_bf16_kernel<<<nx / (4 * 256), 256, 0, stream>>>(x, xb, nx);
  cvt_bf16_kernel<<<nw / (4 * 256), 256, 0, stream>>>(We, wb, nw);
  encode_topk_kernel<<<BT_TOKENS / 32, 128, 0, stream>>>(xb, wb, tv, ti, rowabs);
  decode_kernel<<<BT_TOKENS, 256, 0, stream>>>(Wd, tv, ti, out);
  loss_kernel<<<1, 256, 0, stream>>>(rowabs, out);
}
